// StructuredBackBone8xMRS_22428319220762
// MI455X (gfx1250) — compile-verified, matrix operand order checked by reading, not run on hardware
//
#include <hip/hip_runtime.h>

// ---------------------------------------------------------------------------
// StructuredBackBone8xMRS for MI455X (gfx1250, wave32, WMMA)
//
// Pipeline:
//   1. xyz_kernel: voxel indices (z,y,x) -> world xyz, padded to float4 so the
//      ball-query inner loop is one global_load_b128 per HR point.
//   2. branch_kernel<C,NS,KPAD,N>: per lr point (1 wave each):
//        ball query (ballot + prefix-popcount, in-index-order like top_k(-cand))
//        -> stage [NS x KPAD] grouped f16 features in LDS
//        -> v_wmma_f32_16x16x32_f16 tiles -> max-pool rows -> bias+relu
//   3. out_kernel: [16 x 160] x [160 x 64] WMMA GEMM + BN + ReLU -> d_out
// ---------------------------------------------------------------------------

typedef _Float16 v16h __attribute__((ext_vector_type(16)));
typedef _Float16 h8   __attribute__((ext_vector_type(8)));
typedef float    v8f  __attribute__((ext_vector_type(8)));

#define B_  2
#define M_  4096

__device__ __forceinline__ v16h load_frag(const _Float16* lo, const _Float16* hi) {
  h8 a = *(const h8*)lo;   // K = c0 .. c0+7
  h8 b = *(const h8*)hi;   // K = c0+16 .. c0+23
  return __builtin_shufflevector(a, b, 0,1,2,3,4,5,6,7,8,9,10,11,12,13,14,15);
}

// -------------------------- 1. index -> xyz (float4-padded) ----------------
__global__ void xyz_kernel(const int* __restrict__ idx, float4* __restrict__ xyz,
                           int total, float vx, float vy, float vz) {
  int t = blockIdx.x * blockDim.x + threadIdx.x;
  if (t >= total) return;
  int z = idx[3 * t + 0];
  int y = idx[3 * t + 1];
  int x = idx[3 * t + 2];
  // offset = (0, -40, -3), + half voxel
  float4 p;
  p.x = (float)x * vx + 0.0f  + 0.5f * vx;
  p.y = (float)y * vy - 40.0f + 0.5f * vy;
  p.z = (float)z * vz - 3.0f  + 0.5f * vz;
  p.w = 0.0f;
  xyz[t] = p;
}

// ------------------ 2. fused ball-query + MLP + maxpool --------------------
// One wave per lr point; 4 waves per block share the f16 weight tile.
template <int C, int NS, int KPAD, int N>
__global__ __launch_bounds__(128) void branch_kernel(
    const float4* __restrict__ lr_xyz,  // [B*M]
    const float4* __restrict__ hr_xyz,  // [B*N]
    const float*  __restrict__ hr_feat, // [B*N, C]
    const float*  __restrict__ w,       // [(3+C), 32] row-major
    const float*  __restrict__ bias,    // [32]
    float* __restrict__ fout)           // [B*M, 32]
{
  constexpr int WAVES = 4;
  __shared__ _Float16 wlds[32 * KPAD];            // transposed [o][k], 16B-aligned rows
  __shared__ _Float16 glds[WAVES][NS * KPAD];     // grouped features per wave
  __shared__ int      nbr[WAVES][NS];

  const int lane = threadIdx.x & 31;
  const int wv   = threadIdx.x >> 5;

  // stage weights (transposed, K zero-padded) once per block
  for (int t = threadIdx.x; t < 32 * KPAD; t += 128) {
    int o = t / KPAD, k = t - o * KPAD;
    wlds[t] = (_Float16)((k < 3 + C) ? w[k * 32 + o] : 0.0f);
  }
  __syncthreads();

  const int pt = blockIdx.x * WAVES + wv;          // 0 .. B*M-1
  const int b  = pt >> 12;                          // / M_
  const float4 q = lr_xyz[pt];
  const float4* hxyz = hr_xyz + (size_t)b * N;
  const float*  hf   = hr_feat + (size_t)b * N * C;

  // ---- ball query: first NS indices (ascending) with d2 < 1 ----
  int count = 0;
  for (int base = 0; base < N; base += 32) {
    int j = base + lane;
    float4 p = hxyz[j];                            // single global_load_b128
    float dx = p.x - q.x;
    float dy = p.y - q.y;
    float dz = p.z - q.z;
    bool in = (dx * dx + dy * dy + dz * dz) < 1.0f;   // RADIUS^2
    unsigned m = (unsigned)__ballot(in);
    int pos = count + __popc(m & ((1u << lane) - 1u));
    if (in && pos < NS) nbr[wv][pos] = j;
    count += __popc(m);
    if (count >= NS) break;
  }
  if (count > NS) count = NS;
  // pad with first valid neighbor (or 0 if none)
  int pad = (count > 0) ? nbr[wv][0] : 0;
  for (int s = count + lane; s < NS; s += 32) nbr[wv][s] = pad;
  __syncthreads();

  // ---- stage grouped features: [rel_xyz | feat | zero-pad] as f16 ----
  for (int t = lane; t < NS * KPAD; t += 32) {
    int s = t / KPAD, k = t - s * KPAD;
    int j = nbr[wv][s];
    float v;
    if (k == 0)         v = hxyz[j].x - q.x;
    else if (k == 1)    v = hxyz[j].y - q.y;
    else if (k == 2)    v = hxyz[j].z - q.z;
    else if (k < 3 + C) v = hf[(size_t)j * C + (k - 3)];
    else                v = 0.0f;
    glds[wv][t] = (_Float16)v;
  }
  __syncthreads();

  // ---- WMMA: [NS x KPAD] x [KPAD x 32] with row max-pool ----
  const int khalf = ((lane >> 4) & 1) * 8;
  const int ncol  = lane & 15;
  float cmax0 = -1e30f, cmax1 = -1e30f;
  for (int rt = 0; rt < NS / 16; ++rt) {
    v8f acc0 = {};
    v8f acc1 = {};
    const int mrow = rt * 16 + ncol;               // A row this lane supplies
    const _Float16* grow = &glds[wv][mrow * KPAD];
    for (int kk = 0; kk < KPAD / 32; ++kk) {
      int c0 = kk * 32 + khalf;
      v16h a  = load_frag(grow + c0, grow + c0 + 16);
      v16h b0 = load_frag(&wlds[ncol * KPAD + c0],        &wlds[ncol * KPAD + c0 + 16]);
      v16h b1 = load_frag(&wlds[(ncol + 16) * KPAD + c0], &wlds[(ncol + 16) * KPAD + c0 + 16]);
      acc0 = __builtin_amdgcn_wmma_f32_16x16x32_f16(false, a, false, b0, (short)0, acc0, false, false);
      acc1 = __builtin_amdgcn_wmma_f32_16x16x32_f16(false, a, false, b1, (short)0, acc1, false, false);
    }
    // lane holds column ncol, rows v + 8*(lane>=16): max over the 8 VGPRs
    float m0 = acc0[0], m1 = acc1[0];
    #pragma unroll
    for (int i = 1; i < 8; ++i) { m0 = fmaxf(m0, acc0[i]); m1 = fmaxf(m1, acc1[i]); }
    cmax0 = fmaxf(cmax0, m0);
    cmax1 = fmaxf(cmax1, m1);
  }
  // combine the two row-halves (lane <-> lane^16)
  cmax0 = fmaxf(cmax0, __shfl_xor(cmax0, 16, 32));
  cmax1 = fmaxf(cmax1, __shfl_xor(cmax1, 16, 32));

  if (lane < 16) {
    // max(relu(x+b)) == relu(max(x)+b): bias per column, relu monotone
    fout[(size_t)pt * 32 + ncol]      = fmaxf(cmax0 + bias[ncol],      0.0f);
    fout[(size_t)pt * 32 + 16 + ncol] = fmaxf(cmax1 + bias[ncol + 16], 0.0f);
  }
}

// -------------------- 3. output GEMM [*,160]x[160,64] + BN + ReLU ----------
__global__ __launch_bounds__(128) void out_kernel(
    const float* __restrict__ lr_feat,  // [B*M, 64]
    const float* __restrict__ f14,      // [B*M, 32]
    const float* __restrict__ f24,      // [B*M, 32]
    const float* __restrict__ f34,      // [B*M, 32]
    const float* __restrict__ w_out,    // [160, 64]
    const float* __restrict__ gamma, const float* __restrict__ beta,
    const float* __restrict__ mean,  const float* __restrict__ var,
    float* __restrict__ out)            // [B*M, 64]
{
  constexpr int K = 160, NO = 64, WAVES = 4;
  __shared__ _Float16 wlds[NO * K];          // transposed [o][k], 20 KB
  __shared__ _Float16 glds[WAVES][16 * K];   // 16 rows x 160 per wave, 20 KB

  const int lane = threadIdx.x & 31;
  const int wv   = threadIdx.x >> 5;

  for (int t = threadIdx.x; t < NO * K; t += 128) {
    int o = t / K, k = t - o * K;
    wlds[t] = (_Float16)w_out[k * NO + o];
  }
  __syncthreads();

  const int tile    = blockIdx.x * WAVES + wv;   // 0 .. B*M/16 - 1
  const int rowbase = tile * 16;

  // stage concatenated [lr_feat | f14 | f24 | f34] rows as f16
  for (int t = lane; t < 16 * K; t += 32) {
    int r = t / K, k = t - r * K;
    size_t p = (size_t)(rowbase + r);
    float v;
    if (k < 64)       v = lr_feat[p * 64 + k];
    else if (k < 96)  v = f14[p * 32 + (k - 64)];
    else if (k < 128) v = f24[p * 32 + (k - 96)];
    else              v = f34[p * 32 + (k - 128)];
    glds[wv][t] = (_Float16)v;
  }
  __syncthreads();

  const int khalf = ((lane >> 4) & 1) * 8;
  const int ncol  = lane & 15;
  const _Float16* grow = &glds[wv][ncol * K];    // A row = lane&15

  v8f acc[4] = {};
  for (int kk = 0; kk < K / 32; ++kk) {
    int c0 = kk * 32 + khalf;
    v16h a = load_frag(grow + c0, grow + c0 + 16);
    #pragma unroll
    for (int ct = 0; ct < 4; ++ct) {
      int n = ct * 16 + ncol;
      v16h bf = load_frag(&wlds[n * K + c0], &wlds[n * K + c0 + 16]);
      acc[ct] = __builtin_amdgcn_wmma_f32_16x16x32_f16(false, a, false, bf, (short)0, acc[ct], false, false);
    }
  }

  const int moff = ((lane >> 4) & 1) * 8;
  #pragma unroll
  for (int ct = 0; ct < 4; ++ct) {
    int n = ct * 16 + ncol;
    float sc = gamma[n] * rsqrtf(var[n] + 1e-3f);
    float sh = beta[n] - mean[n] * sc;
    #pragma unroll
    for (int v = 0; v < 8; ++v) {
      int m = moff + v;
      out[(size_t)(rowbase + m) * 64 + n] = fmaxf(acc[ct][v] * sc + sh, 0.0f);
    }
  }
}

// ---------------------------------------------------------------------------
extern "C" void kernel_launch(void* const* d_in, const int* in_sizes, int n_in,
                              void* d_out, int out_size, void* d_ws, size_t ws_size,
                              hipStream_t stream) {
  (void)in_sizes; (void)n_in; (void)out_size; (void)ws_size;
  const int N1 = 16384, N2 = 8192, N3 = 4096;

  const int*   lr_idx   = (const int*)  d_in[0];
  const int*   hr1_idx  = (const int*)  d_in[1];
  const int*   hr2_idx  = (const int*)  d_in[2];
  const int*   hr3_idx  = (const int*)  d_in[3];
  const float* lr_feat  = (const float*)d_in[4];
  const float* hr1_feat = (const float*)d_in[5];
  const float* hr2_feat = (const float*)d_in[6];
  const float* hr3_feat = (const float*)d_in[7];
  const float* w14 = (const float*)d_in[8];
  const float* b14 = (const float*)d_in[9];
  const float* w24 = (const float*)d_in[10];
  const float* b24 = (const float*)d_in[11];
  const float* w34 = (const float*)d_in[12];
  const float* b34 = (const float*)d_in[13];
  const float* w_o = (const float*)d_in[14];
  const float* bng = (const float*)d_in[15];
  const float* bnb = (const float*)d_in[16];
  const float* bnm = (const float*)d_in[17];
  const float* bnv = (const float*)d_in[18];

  // workspace layout: float4 xyz caches + 3 branch outputs (~4 MB)
  float4* lr_xyz  = (float4*)d_ws;
  float4* hr1_xyz = lr_xyz  + (size_t)B_ * M_;
  float4* hr2_xyz = hr1_xyz + (size_t)B_ * N1;
  float4* hr3_xyz = hr2_xyz + (size_t)B_ * N2;
  float*  f14     = (float*)(hr3_xyz + (size_t)B_ * N3);
  float*  f24     = f14     + (size_t)B_ * M_ * 32;
  float*  f34     = f24     + (size_t)B_ * M_ * 32;

  xyz_kernel<<<(B_ * M_ + 255) / 256, 256, 0, stream>>>(lr_idx,  lr_xyz,  B_ * M_, 0.4f,  0.4f,  1.0f);
  xyz_kernel<<<(B_ * N1 + 255) / 256, 256, 0, stream>>>(hr1_idx, hr1_xyz, B_ * N1, 0.05f, 0.05f, 0.1f);
  xyz_kernel<<<(B_ * N2 + 255) / 256, 256, 0, stream>>>(hr2_idx, hr2_xyz, B_ * N2, 0.1f,  0.1f,  0.2f);
  xyz_kernel<<<(B_ * N3 + 255) / 256, 256, 0, stream>>>(hr3_idx, hr3_xyz, B_ * N3, 0.2f,  0.2f,  0.4f);

  // one wave per lr point, 4 waves/block
  branch_kernel<16, 128, 32, 16384><<<B_ * M_ / 4, 128, 0, stream>>>(
      lr_xyz, hr1_xyz, hr1_feat, w14, b14, f14);
  branch_kernel<32, 32, 64, 8192><<<B_ * M_ / 4, 128, 0, stream>>>(
      lr_xyz, hr2_xyz, hr2_feat, w24, b24, f24);
  branch_kernel<64, 16, 96, 4096><<<B_ * M_ / 4, 128, 0, stream>>>(
      lr_xyz, hr3_xyz, hr3_feat, w34, b34, f34);

  // output head: 512 row-tiles of 16 points, 4 waves/block
  out_kernel<<<(B_ * M_ / 16) / 4, 128, 0, stream>>>(
      lr_feat, f14, f24, f34, w_o, bng, bnb, bnm, bnv, (float*)d_out);
}
